// TransformerBlock_58789512347974
// MI455X (gfx1250) — matrix operand order repaired and checked by reading, not compiled, not run on hardware
//
#include <hip/hip_runtime.h>
#include <hip/hip_bf16.h>

// Shapes (fixed by the reference): B=1, S=2048, D=1024, H=16, DH=64, M=4096
#define SEQ 2048
#define DMODEL 1024
#define NHEAD 16
#define DHEAD 64
#define DMLP 4096

typedef __attribute__((ext_vector_type(8)))  float  v8f;
typedef __attribute__((ext_vector_type(16))) __bf16 v16bf;

union Frag16 { v16bf v; unsigned int u[8]; };

__device__ __forceinline__ unsigned short f2bf(float f) {
  unsigned int u = __float_as_uint(f);
  u += 0x7fffu + ((u >> 16) & 1u);   // round-to-nearest-even
  return (unsigned short)(u >> 16);
}

__device__ __forceinline__ float gelu_new(float x) {
  float x3 = x * x * x;
  return 0.5f * x * (1.0f + tanhf(0.7978845608028654f * (x + 0.044715f * x3)));
}

// ---------------------------------------------------------------------------
// TransformerLens LayerNorm: center, divide by RMS(centered), scale+shift.
// ---------------------------------------------------------------------------
__global__ __launch_bounds__(256)
void ln_kernel(const float* __restrict__ x, const float* __restrict__ w,
               const float* __restrict__ b, unsigned short* __restrict__ out) {
  __shared__ float red[256];
  const int row = blockIdx.x, tid = threadIdx.x;
  float v[4];
  float s = 0.f;
  #pragma unroll
  for (int i = 0; i < 4; ++i) { v[i] = x[row * DMODEL + i * 256 + tid]; s += v[i]; }
  red[tid] = s; __syncthreads();
  for (int o = 128; o > 0; o >>= 1) { if (tid < o) red[tid] += red[tid + o]; __syncthreads(); }
  const float mean = red[0] * (1.0f / DMODEL);
  __syncthreads();
  float ss = 0.f;
  #pragma unroll
  for (int i = 0; i < 4; ++i) { v[i] -= mean; ss += v[i] * v[i]; }
  red[tid] = ss; __syncthreads();
  for (int o = 128; o > 0; o >>= 1) { if (tid < o) red[tid] += red[tid + o]; __syncthreads(); }
  const float inv = 1.0f / sqrtf(red[0] * (1.0f / DMODEL) + 1e-5f);
  #pragma unroll
  for (int i = 0; i < 4; ++i) {
    int idx = i * 256 + tid;
    out[row * DMODEL + idx] = f2bf(v[i] * inv * w[idx] + b[idx]);
  }
}

// ---------------------------------------------------------------------------
// B-tile staging: 32(k) x 128(n) fp32 -> bf16, stored transposed [n][k].
// WL=1: W is (H, K, 64) like W_Q/W_K/W_V; WL=0: plain row-major K x N.
// float4 loads never cross a 64-wide head block for aligned groups of 4.
// ---------------------------------------------------------------------------
template <int WL>
__device__ __forceinline__ void stageB(const float* __restrict__ W, int K, int N,
                                       int bn, int k0, int tid,
                                       unsigned short* __restrict__ sB) {
  #pragma unroll
  for (int i = 0; i < 4; ++i) {
    int idx = i * 256 + tid;
    int n4 = idx & 31, kr = idx >> 5;
    int ng = bn + n4 * 4;
    const float* src;
    if (WL) src = &W[(size_t)((ng >> 6) * K + (k0 + kr)) * 64 + (ng & 63)];
    else    src = &W[(size_t)(k0 + kr) * N + ng];
    float4 wv = *(const float4*)src;
    unsigned short* d = &sB[(n4 * 4) * 36 + kr];
    d[0]   = f2bf(wv.x);
    d[36]  = f2bf(wv.y);
    d[72]  = f2bf(wv.z);
    d[108] = f2bf(wv.w);
  }
}

// ---------------------------------------------------------------------------
// WMMA GEMM: C[S x N] = A(bf16)[S x K] * W(fp32->bf16)[K x N] + bias
//   MODE 0: out bf16 [s][n]        MODE 1: out bf16 transposed [n][s]
//   MODE 2: gelu_new -> bf16       MODE 3: +resid -> fp32
// Block tile 64(M) x 128(N), 8 waves, each wave 16x64 (4 accumulators).
// A fragments straight from global; B double-buffered through LDS.
// ---------------------------------------------------------------------------
template <int WL, int MODE>
__global__ __launch_bounds__(256)
void gemm_kernel(const unsigned short* __restrict__ A, int lda,
                 const float* __restrict__ W, int K, int N,
                 const float* __restrict__ bias,
                 unsigned short* __restrict__ outB, float* __restrict__ outF,
                 const float* __restrict__ resid) {
  __shared__ unsigned short sB[2][128 * 36];
  const int tid  = threadIdx.x;
  const int wave = tid >> 5, lane = tid & 31;
  const int cl = lane & 15, hs = lane >> 4;
  const int mt = wave & 3, nt = wave >> 2;
  const int bm = blockIdx.x * 64;
  const int bn = blockIdx.y * 128;

  v8f acc[4];
  #pragma unroll
  for (int c = 0; c < 4; ++c)
    #pragma unroll
    for (int r = 0; r < 8; ++r) acc[c][r] = 0.f;

  const int arow = bm + mt * 16 + cl;
  const int nk = K / 32;

  stageB<WL>(W, K, N, bn, 0, tid, sB[0]);
  __syncthreads();

  for (int ks = 0; ks < nk; ++ks) {
    const int k0 = ks * 32;
    const unsigned short* cur = sB[ks & 1];
    if (ks + 1 < nk)
      stageB<WL>(W, K, N, bn, k0 + 32, tid, sB[(ks + 1) & 1]);

    Frag16 a;
    {
      const unsigned int* ap = (const unsigned int*)(A + (size_t)arow * lda + k0);
      #pragma unroll
      for (int p = 0; p < 8; ++p) {
        int koff = p * 2 + hs * 16;
        a.u[p] = ap[koff >> 1];
      }
    }
    #pragma unroll
    for (int c = 0; c < 4; ++c) {
      Frag16 bfr;
      const int nl = nt * 64 + c * 16 + cl;
      const unsigned int* bp = (const unsigned int*)(cur + nl * 36);
      #pragma unroll
      for (int p = 0; p < 8; ++p) {
        int koff = p * 2 + hs * 16;
        bfr.u[p] = bp[koff >> 1];
      }
      acc[c] = __builtin_amdgcn_wmma_f32_16x16x32_bf16(
          false, a.v, false, bfr.v, (short)0, acc[c], false, false);
    }
    __syncthreads();
  }

  // epilogue (compile-time MODE)
  #pragma unroll
  for (int c = 0; c < 4; ++c) {
    const int ng = bn + nt * 64 + c * 16 + cl;
    const float bv = bias[ng];
    #pragma unroll
    for (int r = 0; r < 8; ++r) {
      const int s = bm + mt * 16 + r + hs * 8;
      float val = acc[c][r] + bv;
      if (MODE == 2) val = gelu_new(val);
      if (MODE == 3)      outF[(size_t)s * N + ng] = val + resid[(size_t)s * N + ng];
      else if (MODE == 1) outB[(size_t)ng * SEQ + s] = f2bf(val);
      else                outB[(size_t)s * N + ng] = f2bf(val);
    }
  }
}

// ---------------------------------------------------------------------------
// Causal flash attention: one (head, 128 query rows) per block, each wave
// owns 16 rows and streams 32-key blocks. q,k: bf16 [s][h*64+e]; vt: bf16
// [h*64+e][s]; z: bf16 [s][h*64+e].
// ---------------------------------------------------------------------------
__global__ __launch_bounds__(256)
void attn_kernel(const unsigned short* __restrict__ q,
                 const unsigned short* __restrict__ k,
                 const unsigned short* __restrict__ vt,
                 unsigned short* __restrict__ z) {
  __shared__ unsigned short sP[8 * 16 * 32];  // per-wave P tile
  const int tid  = threadIdx.x;
  const int wave = tid >> 5, lane = tid & 31;
  const int cl = lane & 15, hs = lane >> 4;
  const int h  = blockIdx.y;
  const int qb = blockIdx.x * 128 + wave * 16;
  unsigned short* Pw = sP + wave * (16 * 32);

  Frag16 qa[2];
  {
    const unsigned int* qp =
        (const unsigned int*)(q + (size_t)(qb + cl) * DMODEL + h * DHEAD);
    #pragma unroll
    for (int t = 0; t < 2; ++t) {
      #pragma unroll
      for (int p = 0; p < 8; ++p) {
        int koff = t * 32 + p * 2 + hs * 16;
        qa[t].u[p] = qp[koff >> 1];
      }
    }
  }

  v8f zacc[4];
  #pragma unroll
  for (int c = 0; c < 4; ++c)
    #pragma unroll
    for (int r = 0; r < 8; ++r) zacc[c][r] = 0.f;
  float m_r[8], l_r[8];
  #pragma unroll
  for (int r = 0; r < 8; ++r) { m_r[r] = -1e30f; l_r[r] = 0.f; }

  const int nkb = (qb + 47) >> 5;
  for (int kb = 0; kb < nkb; ++kb) {
    const int kc = kb * 32;
    v8f sacc[2];
    #pragma unroll
    for (int c = 0; c < 2; ++c)
      #pragma unroll
      for (int r = 0; r < 8; ++r) sacc[c][r] = 0.f;

    #pragma unroll
    for (int cc = 0; cc < 2; ++cc) {
      const unsigned int* kp = (const unsigned int*)(
          k + (size_t)(kc + cc * 16 + cl) * DMODEL + h * DHEAD);
      #pragma unroll
      for (int t = 0; t < 2; ++t) {
        Frag16 bk;
        #pragma unroll
        for (int p = 0; p < 8; ++p) {
          int koff = t * 32 + p * 2 + hs * 16;
          bk.u[p] = kp[koff >> 1];
        }
        sacc[cc] = __builtin_amdgcn_wmma_f32_16x16x32_bf16(
            false, qa[t].v, false, bk.v, (short)0, sacc[cc], false, false);
      }
    }

    #pragma unroll
    for (int r = 0; r < 8; ++r) {
      const int row = qb + r + hs * 8;
      float v0 = sacc[0][r] * 0.125f;
      float v1 = sacc[1][r] * 0.125f;
      if (kc + cl > row)      v0 = -1e5f;
      if (kc + 16 + cl > row) v1 = -1e5f;
      float mx = fmaxf(v0, v1);
      mx = fmaxf(mx, __shfl_xor(mx, 1, 32));
      mx = fmaxf(mx, __shfl_xor(mx, 2, 32));
      mx = fmaxf(mx, __shfl_xor(mx, 4, 32));
      mx = fmaxf(mx, __shfl_xor(mx, 8, 32));
      const float nm = fmaxf(m_r[r], mx);
      const float p0 = __expf(v0 - nm);
      const float p1 = __expf(v1 - nm);
      float ts = p0 + p1;
      ts += __shfl_xor(ts, 1, 32);
      ts += __shfl_xor(ts, 2, 32);
      ts += __shfl_xor(ts, 4, 32);
      ts += __shfl_xor(ts, 8, 32);
      const float sc = __expf(m_r[r] - nm);
      l_r[r] = l_r[r] * sc + ts;
      m_r[r] = nm;
      #pragma unroll
      for (int c = 0; c < 4; ++c) zacc[c][r] *= sc;
      const int rl = r + hs * 8;
      Pw[rl * 32 + cl]      = f2bf(p0);
      Pw[rl * 32 + 16 + cl] = f2bf(p1);
    }
    // per-wave LDS exchange: DS ops are in-order within a wave
    asm volatile("s_wait_dscnt 0" ::: "memory");

    Frag16 pa;
    {
      const unsigned int* pp = (const unsigned int*)(Pw + cl * 32);
      #pragma unroll
      for (int p = 0; p < 8; ++p) {
        int koff = p * 2 + hs * 16;
        pa.u[p] = pp[koff >> 1];
      }
    }
    #pragma unroll
    for (int c = 0; c < 4; ++c) {
      Frag16 bv;
      const unsigned int* vp = (const unsigned int*)(
          vt + (size_t)(h * DHEAD + c * 16 + cl) * SEQ + kc);
      #pragma unroll
      for (int p = 0; p < 8; ++p) {
        int koff = p * 2 + hs * 16;
        bv.u[p] = vp[koff >> 1];
      }
      zacc[c] = __builtin_amdgcn_wmma_f32_16x16x32_bf16(
          false, pa.v, false, bv.v, (short)0, zacc[c], false, false);
    }
  }

  #pragma unroll
  for (int c = 0; c < 4; ++c)
    #pragma unroll
    for (int r = 0; r < 8; ++r) {
      const int sq = qb + r + hs * 8;
      z[(size_t)sq * DMODEL + h * DHEAD + c * 16 + cl] =
          f2bf(zacc[c][r] / l_r[r]);
    }
}

// ---------------------------------------------------------------------------
extern "C" void kernel_launch(void* const* d_in, const int* in_sizes, int n_in,
                              void* d_out, int out_size, void* d_ws, size_t ws_size,
                              hipStream_t stream) {
  (void)in_sizes; (void)n_in; (void)out_size; (void)ws_size;
  const float* resid_pre = (const float*)d_in[0];
  const float* ln1_w = (const float*)d_in[1];
  const float* ln1_b = (const float*)d_in[2];
  const float* W_Q   = (const float*)d_in[3];
  const float* W_K   = (const float*)d_in[4];
  const float* W_V   = (const float*)d_in[5];
  const float* W_O   = (const float*)d_in[6];
  const float* b_Q   = (const float*)d_in[7];
  const float* b_K   = (const float*)d_in[8];
  const float* b_V   = (const float*)d_in[9];
  const float* b_O   = (const float*)d_in[10];
  const float* ln2_w = (const float*)d_in[11];
  const float* ln2_b = (const float*)d_in[12];
  const float* W_in  = (const float*)d_in[13];
  const float* b_in  = (const float*)d_in[14];
  const float* W_out = (const float*)d_in[15];
  const float* b_out = (const float*)d_in[16];

  char* ws = (char*)d_ws;
  unsigned short* x1 = (unsigned short*)(ws + ((size_t)0  << 20)); // 4 MB bf16 LN1(x)
  unsigned short* qf = (unsigned short*)(ws + ((size_t)4  << 20)); // 4 MB bf16 Q
  unsigned short* kf = (unsigned short*)(ws + ((size_t)8  << 20)); // 4 MB bf16 K
  unsigned short* vt = (unsigned short*)(ws + ((size_t)12 << 20)); // 4 MB bf16 V^T
  unsigned short* zf = (unsigned short*)(ws + ((size_t)16 << 20)); // 4 MB bf16 Z
  float*       rmid  = (float*)        (ws + ((size_t)20 << 20)); // 8 MB f32 resid_mid
  unsigned short* x2 = (unsigned short*)(ws + ((size_t)28 << 20)); // 4 MB bf16 LN2(x)
  unsigned short* gf = (unsigned short*)(ws + ((size_t)32 << 20)); // 16 MB bf16 gelu(pre)
  float* out = (float*)d_out;

  ln_kernel<<<SEQ, 256, 0, stream>>>(resid_pre, ln1_w, ln1_b, x1);

  dim3 g1k(SEQ / 64, DMODEL / 128);
  gemm_kernel<1, 0><<<g1k, 256, 0, stream>>>(x1, DMODEL, W_Q, DMODEL, DMODEL, b_Q, qf, nullptr, nullptr);
  gemm_kernel<1, 0><<<g1k, 256, 0, stream>>>(x1, DMODEL, W_K, DMODEL, DMODEL, b_K, kf, nullptr, nullptr);
  gemm_kernel<1, 1><<<g1k, 256, 0, stream>>>(x1, DMODEL, W_V, DMODEL, DMODEL, b_V, vt, nullptr, nullptr);

  attn_kernel<<<dim3(SEQ / 128, NHEAD), 256, 0, stream>>>(qf, kf, vt, zf);

  // attn_out + resid_pre -> resid_mid (fp32)
  gemm_kernel<0, 3><<<g1k, 256, 0, stream>>>(zf, DMODEL, W_O, DMODEL, DMODEL, b_O, nullptr, rmid, resid_pre);

  ln_kernel<<<SEQ, 256, 0, stream>>>(rmid, ln2_w, ln2_b, x2);

  // MLP in: gelu_new fused, bf16 out
  gemm_kernel<0, 2><<<dim3(SEQ / 64, DMLP / 128), 256, 0, stream>>>(x2, DMODEL, W_in, DMODEL, DMLP, b_in, gf, nullptr, nullptr);

  // MLP out: + resid_mid -> final fp32 output
  gemm_kernel<0, 3><<<g1k, 256, 0, stream>>>(gf, DMLP, W_out, DMLP, DMODEL, b_out, nullptr, out, rmid);
}
